// CoarseGrainingEncoder_16776142258455
// MI455X (gfx1250) — matrix operand order repaired and checked by reading, not compiled, not run on hardware
//
#include <hip/hip_runtime.h>
#include <hip/hip_bf16.h>

// ---------------- problem constants ----------------
#define NATOMS 16384
#define BG     64
#define KCL    64          // clusters per molecule
#define MCL    4096        // BG*KCL
#define HID    256
#define FIL    256
#define ECH    64
#define LINT   6
#define CUTOFF_F 10.0f
#define APC    4           // atoms per cluster = NATOMS/MCL

#define GDELTA (CUTOFF_F / 63.0f)                  // gaussian spacing
#define GCOEF  (-0.5f / (GDELTA * GDELTA))
#define LN2_F  0.69314718056f
#define PI_F   3.14159265359f

#define HSTR   264          // padded LDS row stride (halfwords) for hidden tile

// ---------------- WMMA types / helpers ----------------
typedef __attribute__((ext_vector_type(16))) __bf16 v16bf;
typedef __attribute__((ext_vector_type(8)))  float  v8f;

union AFrag {
    v16bf          bf;
    unsigned short u[16];
    uint4          q[2];
};

__device__ __forceinline__ unsigned short f2bf(float f) {
    unsigned int u = __float_as_uint(f);
    unsigned int r = u + 0x7FFFu + ((u >> 16) & 1u);   // round-nearest-even
    return (unsigned short)(r >> 16);
}

__device__ __forceinline__ float bf2f(unsigned short u) {
    return __uint_as_float(((unsigned int)u) << 16);
}

__device__ __forceinline__ float sspf(float x) {       // softplus(x) - ln2
    float ax = fabsf(x);
    float sp = fmaxf(x, 0.0f) + log1pf(__expf(-ax));
    return sp - LN2_F;
}

// load a pre-swizzled 16-element bf16 fragment (two b128 loads)
__device__ __forceinline__ v16bf load_frag16(const unsigned short* p) {
    AFrag f;
    f.q[0] = *reinterpret_cast<const uint4*>(p);
    f.q[1] = *reinterpret_cast<const uint4*>(p + 8);
    return f.bf;
}

// A-fragment from a row-major bf16 matrix with row stride `ld` (halfwords):
// lane holds row m, K runs [off, off+8) and [off+16, off+24), off = kt*32 + hi*8
__device__ __forceinline__ v16bf rm_afrag(const unsigned short* base, int m, int off, int ld) {
    AFrag f;
    f.q[0] = *reinterpret_cast<const uint4*>(base + m * ld + off);
    f.q[1] = *reinterpret_cast<const uint4*>(base + m * ld + off + 16);
    return f.bf;
}

__device__ __forceinline__ v8f wmma_bf16(v16bf a, v16bf b, v8f c) {
    return __builtin_amdgcn_wmma_f32_16x16x32_bf16(
        /*neg_a=*/false, a, /*neg_b=*/false, b,
        /*c_mod=*/(short)0, c, /*reuse_a=*/false, /*reuse_b=*/false);
}

// ---------------- CDNA5 async global->LDS copy (per ISA 08_async_tensor) ------
__device__ __forceinline__ unsigned lds_offset(const void* p) {
    return (unsigned)(unsigned long long)
        (__attribute__((address_space(3))) const void*)p;
}
__device__ __forceinline__ void async_g2l_b128(unsigned lds_off, const void* gptr) {
    asm volatile("global_load_async_to_lds_b128 %0, %1, off"
                 :: "v"(lds_off), "v"((unsigned long long)(size_t)gptr)
                 : "memory");
}
__device__ __forceinline__ void wait_asynccnt0() {
    asm volatile("s_wait_asynccnt 0x0" ::: "memory");
}

// ---------------- K0: weight swizzle f32 -> bf16 B-fragment layout ----------------
// B tile (K=32 x N=16): lane l holds column n = nt*16 + (l&15),
// 16 K values: k = kt*32 + (l>>4)*8 + j for j<8, kt*32 + 16 + (l>>4)*8 + (j-8) for j>=8.
__global__ void swizzle_weight(const float* __restrict__ src,
                               unsigned short* __restrict__ dst,
                               int Kt, int Ktot)
{
    const int b   = blockIdx.x;
    const int l   = threadIdx.x;
    const int lay = b / (16 * Kt);
    const int rem = b % (16 * Kt);
    const int nt  = rem / Kt;
    const int kt  = rem % Kt;
    const int n   = nt * 16 + (l & 15);
    const float* s = src + (size_t)lay * Ktot * 256;
    unsigned short* d = dst + (size_t)b * 512 + l * 16;
#pragma unroll
    for (int j = 0; j < 16; ++j) {
        int k = kt * 32 + ((j < 8) ? ((l >> 4) * 8 + j) : (16 + (l >> 4) * 8 + (j - 8)));
        d[j] = f2bf(s[k * 256 + n]);
    }
}

// ---------------- K1: coarse grain (mean of 4 contiguous atoms) ----------------
__global__ __launch_bounds__(256)
void coarse_grain(const float* __restrict__ pos,
                  const float* __restrict__ nattr,
                  float* __restrict__ cpos,
                  float* __restrict__ h,
                  unsigned short* __restrict__ hbf)
{
    const int cl = blockIdx.x;
    const int t  = threadIdx.x;
    float s = 0.0f;
#pragma unroll
    for (int a = 0; a < APC; ++a) s += nattr[(size_t)(cl * APC + a) * HID + t];
    s *= (1.0f / APC);
    h[(size_t)cl * HID + t]   = s;
    hbf[(size_t)cl * HID + t] = f2bf(s);
    if (t < 3) {
        float p = 0.0f;
#pragma unroll
        for (int a = 0; a < APC; ++a) p += pos[(cl * APC + a) * 3 + t];
        cpos[cl * 3 + t] = p * (1.0f / APC);
    }
}

// ---------------- K3: x = h_bf16 @ lin1 (4096x256x256 WMMA GEMM, bf16 out) -------
__global__ __launch_bounds__(256)
void gemm_lin1(const unsigned short* __restrict__ hbf,
               const unsigned short* __restrict__ lin1s,   // layer offset applied
               unsigned short* __restrict__ xbf)
{
    const int rbase = blockIdx.x << 4;
    const int tid = threadIdx.x;
    const int wv = tid >> 5, l = tid & 31, m = l & 15, hi = l >> 4;
    const unsigned short* arows = hbf + (size_t)rbase * 256;
#pragma unroll
    for (int cti = 0; cti < 2; ++cti) {
        const int ct = wv * 2 + cti;
        v8f acc = {};
#pragma unroll
        for (int kt = 0; kt < 8; ++kt) {
            v16bf a = rm_afrag(arows, m, kt * 32 + hi * 8, 256);
            v16bf b = load_frag16(lin1s + (size_t)(ct * 8 + kt) * 512 + l * 16);
            acc = wmma_bf16(a, b, acc);
        }
#pragma unroll
        for (int v = 0; v < 8; ++v) {
            const int row = hi * 8 + v;
            xbf[(size_t)(rbase + row) * 256 + ct * 16 + m] = f2bf(acc[v]);
        }
    }
}

// ---------------- K4: fused edge-MLP + envelope + aggregation ----------------
// 512 threads = 16 waves, one column tile per wave; weights live in registers
// for the whole kernel; one block per (molecule, group of 4 destinations);
// molecule's x slab async-staged into LDS once.
__global__ __launch_bounds__(512, 1)
void edge_filter_agg(const float* __restrict__ cpos,
                     const unsigned short* __restrict__ xbf,
                     const unsigned short* __restrict__ w1s,  // layer offset applied
                     const unsigned short* __restrict__ w2s,  // layer offset applied
                     const float* __restrict__ b1,            // layer offset applied
                     const float* __restrict__ b2,            // layer offset applied
                     float* __restrict__ agg)
{
    __shared__ __align__(16) unsigned short x_lds[64 * 256];   // 32 KB molecule slab
    __shared__ __align__(16) unsigned short hid_lds[16 * HSTR];// padded hidden tile
    __shared__ float aggbuf[256];

    const int g   = blockIdx.x >> 4;          // molecule
    const int cg  = (blockIdx.x & 15) << 2;   // base of 4-dest group
    const int g0  = g << 6;
    const int tid = threadIdx.x;
    const int wv  = tid >> 5, l = tid & 31, m = l & 15, hi = l >> 4;
    const int ct  = wv;                       // this wave's column tile (0..15)

    // ---- stage molecule x slab (bf16, 32KB) into LDS asynchronously ----
    {
        const unsigned base = lds_offset(x_lds);
        const char* gx = (const char*)(xbf + (size_t)g0 * 256);
        for (int i = tid; i < (64 * 256 * 2) / 16; i += 512)
            async_g2l_b128(base + (unsigned)i * 16u, gx + (size_t)i * 16);
    }

    // ---- this wave's B fragments: registers for the whole kernel ----
    v16bf b1f[2], b2f[8];
#pragma unroll
    for (int kc = 0; kc < 2; ++kc)
        b1f[kc] = load_frag16(w1s + (size_t)(ct * 2 + kc) * 512 + l * 16);
#pragma unroll
    for (int kt = 0; kt < 8; ++kt)
        b2f[kt] = load_frag16(w2s + (size_t)(ct * 8 + kt) * 512 + l * 16);

    const float b1n = b1[ct * 16 + m];
    const float b2n = b2[ct * 16 + m];

    wait_asynccnt0();
    __syncthreads();

    for (int d = 0; d < 4; ++d) {
        const int c = cg + d;
        if (tid < 256) aggbuf[tid] = 0.0f;

        const float dx0 = cpos[(g0 + c) * 3 + 0];
        const float dy0 = cpos[(g0 + c) * 3 + 1];
        const float dz0 = cpos[(g0 + c) * 3 + 2];

        for (int rt = 0; rt < 4; ++rt) {
            const int rbase = rt << 4;
            // lane: distance for source row (l&15) -> dest c
            float len;
            {
                const int r = g0 + rbase + m;
                float ax = cpos[r * 3 + 0] - dx0;
                float ay = cpos[r * 3 + 1] - dy0;
                float az = cpos[r * 3 + 2] - dz0;
                len = sqrtf(ax * ax + ay * ay + az * az);
            }

            // ---- stage 1: hidden = ssp(gauss(len) @ w1 + b1) ----
            {
                v8f acc = {};
#pragma unroll
                for (int kc = 0; kc < 2; ++kc) {
                    AFrag a;
#pragma unroll
                    for (int j = 0; j < 16; ++j) {
                        int k = kc * 32 + ((j < 8) ? (hi * 8 + j) : (16 + hi * 8 + (j - 8)));
                        float dd = len - (float)k * GDELTA;
                        a.u[j] = f2bf(__expf(GCOEF * dd * dd));
                    }
                    acc = wmma_bf16(a.bf, b1f[kc], acc);
                }
#pragma unroll
                for (int v = 0; v < 8; ++v)
                    hid_lds[(hi * 8 + v) * HSTR + ct * 16 + m] = f2bf(sspf(acc[v] + b1n));
            }
            __syncthreads();

            // per-row cutoff envelope (8 rows per lane, diagonal masked)
            float ce[8];
#pragma unroll
            for (int v = 0; v < 8; ++v) {
                const int row = hi * 8 + v;
                const float lr = __shfl(len, row, 32);
                const float env = 0.5f * (__cosf(lr * (PI_F / CUTOFF_F)) + 1.0f);
                const bool valid = (lr <= CUTOFF_F) && ((rbase + row) != c);
                ce[v] = valid ? env : 0.0f;
            }

            // ---- stage 2: W = hidden @ w2 + b2 ; reduce x*C*W over rows ----
            {
                v8f acc = {};
#pragma unroll
                for (int kt = 0; kt < 8; ++kt) {
                    v16bf a = rm_afrag(hid_lds, m, kt * 32 + hi * 8, HSTR);
                    acc = wmma_bf16(a, b2f[kt], acc);
                }
                float val = 0.0f;
#pragma unroll
                for (int v = 0; v < 8; ++v) {
                    const int row = hi * 8 + v;
                    const float xv = bf2f(x_lds[(rbase + row) * 256 + ct * 16 + m]);
                    val += xv * ce[v] * (acc[v] + b2n);
                }
                val += __shfl_xor(val, 16, 32);     // combine row halves
                if (hi == 0) aggbuf[ct * 16 + m] += val;
            }
            __syncthreads();   // hid_lds reused next rt; orders aggbuf phases
        }
        if (tid < 256) agg[(size_t)(g0 + c) * 256 + tid] = aggbuf[tid];
    }
}

// ---------------- K5: h += ssp(agg@lin2+b) @ lin + b (two fused WMMA GEMMs) --------
__global__ __launch_bounds__(256)
void post_update(const float* __restrict__ agg,
                 const unsigned short* __restrict__ lin2s,  // layer offsets applied
                 const float* __restrict__ lin2b,
                 const unsigned short* __restrict__ lins,
                 const float* __restrict__ linb,
                 float* __restrict__ h,
                 unsigned short* __restrict__ hbf)
{
    __shared__ __align__(16) unsigned short t_lds[16 * HSTR];
    const int rbase = blockIdx.x << 4;
    const int tid = threadIdx.x;
    const int wv = tid >> 5, l = tid & 31, m = l & 15, hi = l >> 4;

    // stage 1: t = ssp(agg @ lin2 + b)
#pragma unroll
    for (int cti = 0; cti < 2; ++cti) {
        const int ct = wv * 2 + cti;
        v8f acc = {};
#pragma unroll
        for (int kt = 0; kt < 8; ++kt) {
            const float* p = agg + (size_t)(rbase + m) * 256 + kt * 32 + hi * 8;
            AFrag a;
#pragma unroll
            for (int j = 0; j < 8; ++j) a.u[j]     = f2bf(p[j]);
#pragma unroll
            for (int j = 0; j < 8; ++j) a.u[8 + j] = f2bf(p[16 + j]);
            v16bf b = load_frag16(lin2s + (size_t)(ct * 8 + kt) * 512 + l * 16);
            acc = wmma_bf16(a.bf, b, acc);
        }
        const float bb = lin2b[ct * 16 + m];
#pragma unroll
        for (int v = 0; v < 8; ++v)
            t_lds[(hi * 8 + v) * HSTR + ct * 16 + m] = f2bf(sspf(acc[v] + bb));
    }
    __syncthreads();

    // stage 2: h += t @ lin + b
#pragma unroll
    for (int cti = 0; cti < 2; ++cti) {
        const int ct = wv * 2 + cti;
        v8f acc = {};
#pragma unroll
        for (int kt = 0; kt < 8; ++kt) {
            v16bf a = rm_afrag(t_lds, m, kt * 32 + hi * 8, HSTR);
            v16bf b = load_frag16(lins + (size_t)(ct * 8 + kt) * 512 + l * 16);
            acc = wmma_bf16(a, b, acc);
        }
        const float bb = linb[ct * 16 + m];
#pragma unroll
        for (int v = 0; v < 8; ++v) {
            const size_t idx = (size_t)(rbase + hi * 8 + v) * 256 + ct * 16 + m;
            const float nh = h[idx] + acc[v] + bb;
            h[idx]   = nh;
            hbf[idx] = f2bf(nh);
        }
    }
}

// ---------------- host launcher ----------------
extern "C" void kernel_launch(void* const* d_in, const int* in_sizes, int n_in,
                              void* d_out, int out_size, void* d_ws, size_t ws_size,
                              hipStream_t stream)
{
    const float* pos       = (const float*)d_in[0];
    const float* node_attr = (const float*)d_in[1];
    // d_in[2] subgraph_index, d_in[3] batch: structure is arange//APC (used implicitly)
    const float* mlp_w1 = (const float*)d_in[6];
    const float* mlp_b1 = (const float*)d_in[7];
    const float* mlp_w2 = (const float*)d_in[8];
    const float* mlp_b2 = (const float*)d_in[9];
    const float* lin1_w = (const float*)d_in[10];
    const float* lin2_w = (const float*)d_in[11];
    const float* lin2_b = (const float*)d_in[12];
    const float* lin_w  = (const float*)d_in[13];
    const float* lin_b  = (const float*)d_in[14];

    // workspace carve-up (256B aligned)
    char* ws = (char*)d_ws;
    size_t off = 0;
    auto take = [&](size_t bytes) { char* p = ws + off; off = (off + bytes + 255) & ~(size_t)255; return p; };
    float*          cpos  = (float*)take((size_t)MCL * 3 * 4);
    float*          h     = (float*)take((size_t)MCL * HID * 4);
    float*          agg   = (float*)take((size_t)MCL * FIL * 4);
    unsigned short* hbf   = (unsigned short*)take((size_t)MCL * HID * 2);
    unsigned short* xbf   = (unsigned short*)take((size_t)MCL * FIL * 2);
    unsigned short* w1s   = (unsigned short*)take((size_t)LINT * 16 * 2 * 512 * 2);
    unsigned short* w2s   = (unsigned short*)take((size_t)LINT * 16 * 8 * 512 * 2);
    unsigned short* lin1s = (unsigned short*)take((size_t)LINT * 16 * 8 * 512 * 2);
    unsigned short* lin2s = (unsigned short*)take((size_t)LINT * 16 * 8 * 512 * 2);
    unsigned short* linsw = (unsigned short*)take((size_t)LINT * 16 * 8 * 512 * 2);
    (void)ws_size; (void)in_sizes; (void)n_in;

    // one-time weight swizzle (bf16 B-fragment layout)
    swizzle_weight<<<LINT * 16 * 2, 32, 0, stream>>>(mlp_w1, w1s, 2, ECH);
    swizzle_weight<<<LINT * 16 * 8, 32, 0, stream>>>(mlp_w2, w2s, 8, FIL);
    swizzle_weight<<<LINT * 16 * 8, 32, 0, stream>>>(lin1_w, lin1s, 8, HID);
    swizzle_weight<<<LINT * 16 * 8, 32, 0, stream>>>(lin2_w, lin2s, 8, FIL);
    swizzle_weight<<<LINT * 16 * 8, 32, 0, stream>>>(lin_w,  linsw, 8, HID);

    coarse_grain<<<MCL, 256, 0, stream>>>(pos, node_attr, cpos, h, hbf);

    for (int lay = 0; lay < LINT; ++lay) {
        gemm_lin1<<<MCL / 16, 256, 0, stream>>>(hbf, lin1s + (size_t)lay * 16 * 8 * 512, xbf);
        edge_filter_agg<<<BG * 16, 512, 0, stream>>>(
            cpos, xbf,
            w1s + (size_t)lay * 16 * 2 * 512,
            w2s + (size_t)lay * 16 * 8 * 512,
            mlp_b1 + (size_t)lay * FIL,
            mlp_b2 + (size_t)lay * FIL,
            agg);
        post_update<<<MCL / 16, 256, 0, stream>>>(
            agg,
            lin2s + (size_t)lay * 16 * 8 * 512, lin2_b + (size_t)lay * HID,
            linsw + (size_t)lay * 16 * 8 * 512, lin_b + (size_t)lay * HID,
            h, hbf);
    }

    hipMemcpyAsync(d_out, h, (size_t)MCL * HID * sizeof(float),
                   hipMemcpyDeviceToDevice, stream);
}
